// Sinkhorn_20658792694050
// MI455X (gfx1250) — matrix operand order repaired and checked by reading, not compiled, not run on hardware
//
#include <hip/hip_runtime.h>
#include <stdint.h>

#define EPSV 1e-12f

constexpr int NB = 64;
constexpr int L1 = 1024;
constexpr int L2 = 1024;
constexpr int DD = 256;

typedef __attribute__((ext_vector_type(16))) __bf16        bf16x16;
typedef __attribute__((ext_vector_type(8)))  __bf16        bf16x8;
typedef __attribute__((ext_vector_type(8)))  float         f32x8;
typedef __attribute__((ext_vector_type(4)))  float         f32x4;
typedef __attribute__((ext_vector_type(4)))  unsigned int  u32x4;

// fp32 -> bf16 round-to-nearest-even (bitwise)
__device__ __forceinline__ unsigned short f2bf(float f) {
  unsigned int u = __float_as_uint(f);
  u += 0x7FFFu + ((u >> 16) & 1u);
  return (unsigned short)(u >> 16);
}
__device__ __forceinline__ float bf2f(unsigned short h) {
  return __uint_as_float(((unsigned int)h) << 16);
}

// Build a 16-element bf16 fragment from two 16B LDS chunks.
__device__ __forceinline__ bf16x16 frag16(const unsigned short* p0,
                                          const unsigned short* p1) {
  u32x4 a = *(const u32x4*)p0;
  u32x4 b = *(const u32x4*)p1;
  bf16x8 x = __builtin_bit_cast(bf16x8, a);
  bf16x8 y = __builtin_bit_cast(bf16x8, b);
  bf16x16 r;
#pragma unroll
  for (int e = 0; e < 8; ++e) { r[e] = x[e]; r[e + 8] = y[e]; }
  return r;
}

// ---------------------------------------------------------------------------
// Kernel 0: per-row L2 norms (8 rows per 256-thread block, one wave per row)
// ---------------------------------------------------------------------------
__global__ __launch_bounds__(256) void row_norms(const float* __restrict__ x,
                                                 float* __restrict__ out) {
  int row  = blockIdx.x * 8 + (threadIdx.x >> 5);
  int lane = threadIdx.x & 31;
  const float* p = x + (size_t)row * DD;
  float s = 0.f;
#pragma unroll
  for (int k = lane; k < DD; k += 32) { float f = p[k]; s += f * f; }
#pragma unroll
  for (int off = 16; off; off >>= 1) s += __shfl_xor(s, off, 32);
  if (lane == 0) out[row] = sqrtf(s);
}

__global__ void fill_ones(float* __restrict__ u) {
  u[blockIdx.x * 256 + threadIdx.x] = 1.0f;
}

// ---------------------------------------------------------------------------
// Kernel 1: cost matrix M = 1 - <x1,x2>/(|x1||x2|+eps) via split-bf16 WMMA.
// Block tile 128x128, 8 waves (4 x 2), each wave 2x4 tiles of 16x16.
// dot(a,b) ~= a_hi*b_hi + a_hi*b_lo + a_lo*b_hi  (3 bf16 WMMAs, ~fp32 accuracy)
// ---------------------------------------------------------------------------
__global__ __launch_bounds__(256) void cost_gemm(const float* __restrict__ x1,
                                                 const float* __restrict__ x2,
                                                 const float* __restrict__ n1,
                                                 const float* __restrict__ n2,
                                                 float* __restrict__ Mo) {
  constexpr int KC  = 32;
  constexpr int LDK = KC + 8;  // 40 ushort pitch: conflict-free & 16B-aligned
  __shared__ __align__(16) unsigned short Ahi[128][LDK];
  __shared__ __align__(16) unsigned short Alo[128][LDK];
  __shared__ __align__(16) unsigned short Bhi[128][LDK];
  __shared__ __align__(16) unsigned short Blo[128][LDK];

  const int b    = blockIdx.z;
  const int i0   = blockIdx.x * 128;
  const int j0   = blockIdx.y * 128;
  const int t    = threadIdx.x;
  const int lane = t & 31;
  const int wid  = t >> 5;
  const int wm   = wid & 3;    // wave row   (32 rows each)
  const int wn   = wid >> 2;   // wave col   (64 cols each)
  const int r16  = lane & 15;
  const int hh   = lane >> 4;  // half-wave id

  const float* Ab = x1 + ((size_t)b * L1 + i0) * DD;
  const float* Bb = x2 + ((size_t)b * L2 + j0) * DD;

  f32x8 acc[2][4] = {};

  const int lrow = t >> 1;          // 0..127
  const int lk   = (t & 1) * 16;    // 0 or 16

  for (int kc = 0; kc < DD; kc += KC) {
    // ---- stage 128x32 chunks of A and B to LDS, split into hi/lo bf16 ----
#pragma unroll
    for (int q = 0; q < 4; ++q) {
      f32x4 fa = *(const f32x4*)(Ab + (size_t)lrow * DD + kc + lk + q * 4);
      f32x4 fb = *(const f32x4*)(Bb + (size_t)lrow * DD + kc + lk + q * 4);
#pragma unroll
      for (int e = 0; e < 4; ++e) {
        unsigned short ha = f2bf(fa[e]);
        Ahi[lrow][lk + q * 4 + e] = ha;
        Alo[lrow][lk + q * 4 + e] = f2bf(fa[e] - bf2f(ha));
        unsigned short hb = f2bf(fb[e]);
        Bhi[lrow][lk + q * 4 + e] = hb;
        Blo[lrow][lk + q * 4 + e] = f2bf(fb[e] - bf2f(hb));
      }
    }
    __syncthreads();

    // ---- load fragments per ISA 16-bit A/B layouts ----
    bf16x16 ah[2], al[2], bh[4], bl[4];
#pragma unroll
    for (int m = 0; m < 2; ++m) {
      int r = wm * 32 + m * 16 + r16;
      ah[m] = frag16(&Ahi[r][hh * 8], &Ahi[r][16 + hh * 8]);
      al[m] = frag16(&Alo[r][hh * 8], &Alo[r][16 + hh * 8]);
    }
#pragma unroll
    for (int n = 0; n < 4; ++n) {
      int c = wn * 64 + n * 16 + r16;
      bh[n] = frag16(&Bhi[c][hh * 8], &Bhi[c][16 + hh * 8]);
      bl[n] = frag16(&Blo[c][hh * 8], &Blo[c][16 + hh * 8]);
    }
    __syncthreads();

    // ---- 24 WMMAs per k-step per wave ----
#pragma unroll
    for (int m = 0; m < 2; ++m)
#pragma unroll
      for (int n = 0; n < 4; ++n) {
        acc[m][n] = __builtin_amdgcn_wmma_f32_16x16x32_bf16(
            false, ah[m], false, bh[n], (short)0, acc[m][n], false, false);
        acc[m][n] = __builtin_amdgcn_wmma_f32_16x16x32_bf16(
            false, ah[m], false, bl[n], (short)0, acc[m][n], false, false);
        acc[m][n] = __builtin_amdgcn_wmma_f32_16x16x32_bf16(
            false, al[m], false, bh[n], (short)0, acc[m][n], false, false);
      }
  }

  // ---- epilogue: M = 1 - dot/(|a||b|+eps)  (C/D layout: row=v+8*hh, col=r16)
#pragma unroll
  for (int n = 0; n < 4; ++n) {
    int col  = j0 + wn * 64 + n * 16 + r16;
    float nb = n2[b * L2 + col];
#pragma unroll
    for (int m = 0; m < 2; ++m) {
#pragma unroll
      for (int vv = 0; vv < 8; ++vv) {
        int row  = i0 + wm * 32 + m * 16 + hh * 8 + vv;
        float na = n1[b * L1 + row];
        float d  = acc[m][n][vv];
        Mo[((size_t)b * L1 + row) * L2 + col] = 1.0f - d / (na * nb + EPSV);
      }
    }
  }
}

// ---------------------------------------------------------------------------
// Sinkhorn pass A: v_j = w2 / (sum_i exp(-M[i,j]) u_i + eps)   (column pass)
// Grid (16 j-chunks of 64, NB). Coalesced: 64 consecutive j per quarter-block.
// ---------------------------------------------------------------------------
__global__ __launch_bounds__(256) void sink_colpass(const float* __restrict__ Mh,
                                                    const float* __restrict__ u,
                                                    float* __restrict__ v) {
  int b  = blockIdx.y;
  int jl = threadIdx.x & 63;
  int j  = blockIdx.x * 64 + jl;
  int ig = threadIdx.x >> 6;  // 0..3
  const float* Mb = Mh + (size_t)b * L1 * L2;
  const float* ub = u + b * L1;
  float s = 0.f;
  for (int i = ig; i < L1; i += 4)
    s += __expf(-__builtin_nontemporal_load(&Mb[(size_t)i * L2 + j])) * ub[i];
  __shared__ float red[4][64];
  red[ig][jl] = s;
  __syncthreads();
  if (threadIdx.x < 64) {
    float tt = red[0][threadIdx.x] + red[1][threadIdx.x] +
               red[2][threadIdx.x] + red[3][threadIdx.x];
    v[b * L2 + blockIdx.x * 64 + threadIdx.x] = (1.0f / L2) / (tt + EPSV);
  }
}

// ---------------------------------------------------------------------------
// Sinkhorn pass B: u_i = w1 / (sum_j exp(-M[i,j]) v_j + eps)   (row pass)
// Grid (128 i-chunks of 8 rows, NB). One wave per row, 128B coalesced reads.
// ---------------------------------------------------------------------------
__global__ __launch_bounds__(256) void sink_rowpass(const float* __restrict__ Mh,
                                                    const float* __restrict__ v,
                                                    float* __restrict__ u) {
  int b    = blockIdx.y;
  int wid  = threadIdx.x >> 5;
  int lane = threadIdx.x & 31;
  int i    = blockIdx.x * 8 + wid;
  const float* Mr = Mh + ((size_t)b * L1 + i) * L2;
  const float* vb = v + b * L2;
  float s = 0.f;
  for (int j = lane; j < L2; j += 32)
    s += __expf(-__builtin_nontemporal_load(&Mr[j])) * vb[j];
#pragma unroll
  for (int off = 16; off; off >>= 1) s += __shfl_xor(s, off, 32);
  if (lane == 0) u[b * L1 + i] = (1.0f / L1) / (s + EPSV);
}

// ---------------------------------------------------------------------------
// Per-batch min/max of G = u_i exp(-M) v_j : stage 1 (fixed-order, no atomics)
// ---------------------------------------------------------------------------
__global__ __launch_bounds__(256) void gminmax_part(const float* __restrict__ Mh,
                                                    const float* __restrict__ u,
                                                    const float* __restrict__ v,
                                                    float* __restrict__ pmin,
                                                    float* __restrict__ pmax) {
  int b = blockIdx.y;
  const float* Mb = Mh + (size_t)b * L1 * L2;
  const float* ub = u + b * L1;
  const float* vb = v + b * L2;
  size_t start = (size_t)blockIdx.x * 32768 + threadIdx.x;
  float mn = 3.4e38f, mx = -3.4e38f;
  for (int it = 0; it < 128; ++it) {
    size_t idx = start + (size_t)it * 256;
    float m = __builtin_nontemporal_load(&Mb[idx]);
    float g = ub[idx >> 10] * __expf(-m) * vb[idx & 1023];
    mn = fminf(mn, g);
    mx = fmaxf(mx, g);
  }
#pragma unroll
  for (int off = 16; off; off >>= 1) {
    mn = fminf(mn, __shfl_xor(mn, off, 32));
    mx = fmaxf(mx, __shfl_xor(mx, off, 32));
  }
  __shared__ float smn[8], smx[8];
  int lane = threadIdx.x & 31, wid = threadIdx.x >> 5;
  if (lane == 0) { smn[wid] = mn; smx[wid] = mx; }
  __syncthreads();
  if (threadIdx.x == 0) {
    float a = smn[0], c = smx[0];
#pragma unroll
    for (int w = 1; w < 8; ++w) { a = fminf(a, smn[w]); c = fmaxf(c, smx[w]); }
    pmin[b * 32 + blockIdx.x] = a;
    pmax[b * 32 + blockIdx.x] = c;
  }
}

__global__ void gminmax_final(const float* __restrict__ pmin,
                              const float* __restrict__ pmax,
                              float* __restrict__ gmin,
                              float* __restrict__ gmax) {
  int b = blockIdx.x, lane = threadIdx.x;  // 32 threads
  float mn = pmin[b * 32 + lane];
  float mx = pmax[b * 32 + lane];
#pragma unroll
  for (int off = 16; off; off >>= 1) {
    mn = fminf(mn, __shfl_xor(mn, off, 32));
    mx = fmaxf(mx, __shfl_xor(mx, off, 32));
  }
  if (lane == 0) { gmin[b] = mn; gmax[b] = mx; }
}

// ---------------------------------------------------------------------------
// Finalize in place: out = ((G - Gmin)/(Gmax - Gmin + eps)) * M, G recomputed.
// ---------------------------------------------------------------------------
__global__ __launch_bounds__(256) void finalize(float* __restrict__ out,
                                                const float* __restrict__ u,
                                                const float* __restrict__ v,
                                                const float* __restrict__ gmin,
                                                const float* __restrict__ gmax) {
  size_t idx = ((size_t)blockIdx.x * 256 + threadIdx.x) * 4;
  int b = (int)(idx >> 20);
  int i = (int)((idx >> 10) & 1023);
  int j = (int)(idx & 1023);
  float um = u[b * L1 + i];
  float gm = gmin[b];
  float sc = 1.0f / (gmax[b] - gm + EPSV);
  f32x4 m  = *(f32x4*)(out + idx);
  f32x4 r;
#pragma unroll
  for (int e = 0; e < 4; ++e) {
    float g = um * __expf(-m[e]) * v[b * L2 + j + e];
    r[e] = (g - gm) * sc * m[e];
  }
  *(f32x4*)(out + idx) = r;
}

// ---------------------------------------------------------------------------
extern "C" void kernel_launch(void* const* d_in, const int* in_sizes, int n_in,
                              void* d_out, int out_size, void* d_ws, size_t ws_size,
                              hipStream_t stream) {
  (void)in_sizes; (void)n_in; (void)out_size; (void)ws_size;
  const float* x1 = (const float*)d_in[0];
  const float* x2 = (const float*)d_in[1];
  float* out = (float*)d_out;

  float* ws   = (float*)d_ws;
  float* n1   = ws;                 // NB*L1
  float* n2   = n1 + NB * L1;       // NB*L2
  float* u    = n2 + NB * L2;       // NB*L1
  float* v    = u + NB * L1;        // NB*L2
  float* pmin = v + NB * L2;        // NB*32
  float* pmax = pmin + NB * 32;     // NB*32
  float* gmn  = pmax + NB * 32;     // NB
  float* gmx  = gmn + NB;           // NB

  row_norms<<<NB * L1 / 8, 256, 0, stream>>>(x1, n1);
  row_norms<<<NB * L2 / 8, 256, 0, stream>>>(x2, n2);
  fill_ones<<<NB * L1 / 256, 256, 0, stream>>>(u);

  cost_gemm<<<dim3(L1 / 128, L2 / 128, NB), 256, 0, stream>>>(x1, x2, n1, n2, out);

  for (int it = 0; it < 10; ++it) {
    sink_colpass<<<dim3(L2 / 64, NB), 256, 0, stream>>>(out, u, v);
    sink_rowpass<<<dim3(L1 / 8, NB), 256, 0, stream>>>(out, v, u);
  }

  gminmax_part<<<dim3(32, NB), 256, 0, stream>>>(out, u, v, pmin, pmax);
  gminmax_final<<<NB, 32, 0, stream>>>(pmin, pmax, gmn, gmx);
  finalize<<<(size_t)NB * L1 * L2 / 1024, 256, 0, stream>>>(out, u, v, gmn, gmx);
}
